// VolumeSDF_14362370638483
// MI455X (gfx1250) — matrix operand order repaired and checked by reading, not compiled, not run on hardware
//
#include <hip/hip_runtime.h>
#include <hip/hip_bf16.h>
#include <math.h>

typedef __attribute__((ext_vector_type(16))) _Float16 v16h;
typedef __attribute__((ext_vector_type(8)))  float    v8f;

#define TBL_SIZE (1u << 19)
#define TBL_MASK (TBL_SIZE - 1u)

union Frag {
    v16h h;
    uint4 q[2];
};

// softplus(100x) WITHOUT the final /100 (folded into w2 weights).
// = max(t,0) + ln2 * log2(1 + 2^(-|t|*log2e)), t = 100x.
// Log argument is in [1,2] -> raw v_log_f32 / v_exp_f32 are safe (no denorm/inf fixup).
__device__ __forceinline__ float softplus100_raw(float x) {
    const float t = 100.0f * x;
    const float e = __builtin_amdgcn_exp2f(fabsf(t) * -1.4426950408889634f);
    const float lg = __builtin_amdgcn_logf(1.0f + e);       // log2(1+e) in [0,1]
    return fmaf(0.69314718055994531f, lg, fmaxf(t, 0.0f));
}

__global__ __launch_bounds__(256) void volumesdf_kernel(
    const float* __restrict__ points,  // [N,3]
    const float* __restrict__ table,   // [16, 2^19, 2]
    const float* __restrict__ v1,      // [64,35]
    const float* __restrict__ g1,      // [64]
    const float* __restrict__ b1,      // [64]
    const float* __restrict__ v2,      // [1,64]
    const float* __restrict__ g2,      // [1]
    const float* __restrict__ b2,      // [1]
    float* __restrict__ out,           // [N]
    long long N)
{
    // LDS: W1 (f16, 64 rows x 72-half stride), per-wave enc staging (8 waves x 16 rows x 72)
    __shared__ __align__(16) _Float16 w1s[64 * 72];
    __shared__ __align__(16) _Float16 stage[8 * 16 * 72];
    __shared__ float b1s[64];
    __shared__ float w2s[64];

    const int tid  = threadIdx.x;
    const int lane = tid & 31;
    const int wave = tid >> 5;
    const int m    = lane & 15;   // point within wave tile / output column within tile
    const int hh   = lane >> 4;   // half selector

    // ---------- per-block weight prep (deterministic, redundant across blocks) ----------
    if (tid < 64) {
        const int j = tid;
        float ss = 0.0f;
        for (int k = 0; k < 35; ++k) { float v = v1[j * 35 + k]; ss += v * v; }
        const float sc = g1[j] / sqrtf(ss);
        // column permutation: k' in [0,32) <- enc col 3+k' (hash feats), k' 32..34 <- coords, rest 0
        for (int kp = 0; kp < 64; ++kp) {
            float v;
            if (kp < 32)      v = v1[j * 35 + 3 + kp] * sc;
            else if (kp < 35) v = v1[j * 35 + (kp - 32)] * sc;
            else              v = 0.0f;
            w1s[j * 72 + kp] = (_Float16)v;
        }
        b1s[j] = b1[j];
    } else if (tid == 64) {
        float ss = 0.0f;
        for (int j = 0; j < 64; ++j) { float v = v2[j]; ss += v * v; }
        const float sc = g2[0] / sqrtf(ss) * 0.01f;   // fold softplus /100 here
        for (int j = 0; j < 64; ++j) w2s[j] = v2[j] * sc;
    }

    // ---------- hashgrid encode: stage 16 points x 64(padded K) halfs per wave ----------
    const long long tileBase = (long long)blockIdx.x * 128 + (long long)wave * 16;
    long long gp = tileBase + m;
    if (gp >= N) gp = N - 1;  // clamp for safety (N is a multiple of 128 in practice)

    const float p0 = points[gp * 3 + 0];
    const float p1 = points[gp * 3 + 1];
    const float p2 = points[gp * 3 + 2];

    _Float16* myStage = stage + wave * (16 * 72);
    unsigned int* myStageU = (unsigned int*)myStage;

    if (hh == 0) {
        myStage[m * 72 + 32] = (_Float16)(p0 * 2.0f - 1.0f);
        myStage[m * 72 + 33] = (_Float16)(p1 * 2.0f - 1.0f);
        myStage[m * 72 + 34] = (_Float16)(p2 * 2.0f - 1.0f);
        myStage[m * 72 + 35] = (_Float16)0.0f;
        for (int k = 36; k < 64; k += 2) myStageU[(m * 72 + k) >> 1] = 0u;
    }

    // per-level scale: 32 * s^l - 1 (s = 1.3195079107728942 = 4^(1/5)).
    // Level mapping l = 2*li + hh keeps the dense/hash predicate uniform across the wave
    // (levels 0..3 dense, 4..15 hashed).
    const double S  = 1.3195079107728942;
    const double S2 = S * S;
    double sp = hh ? S : 1.0;

    const float2* tbl2 = (const float2*)table;

    for (int li = 0; li < 8; ++li) {
        const int l = 2 * li + hh;
        const double scaled = 32.0 * sp - 1.0;
        const float scale = (float)scaled;
        const int res = (int)ceil(scaled) + 1;
        const bool dense = ((long long)res * res * res) <= (long long)TBL_SIZE;

        // pos = p*scale + 0.5 without FMA contraction (match XLA mul-then-add)
        const float fx = __fadd_rn(__fmul_rn(p0, scale), 0.5f);
        const float fy = __fadd_rn(__fmul_rn(p1, scale), 0.5f);
        const float fz = __fadd_rn(__fmul_rn(p2, scale), 0.5f);
        const float x0 = floorf(fx), y0 = floorf(fy), z0 = floorf(fz);
        const float wx = fx - x0, wy = fy - y0, wz = fz - z0;
        const unsigned int ix = (unsigned int)x0;
        const unsigned int iy = (unsigned int)y0;
        const unsigned int iz = (unsigned int)z0;

        // corner weights as explicit products: cw[c] = pxy[bi*2+bj] * pz[bk]
        const float pxm = 1.0f - wx, pym = 1.0f - wy, pzm = 1.0f - wz;
        float pxy[4];
        pxy[0] = pxm * pym;  // bi=0,bj=0
        pxy[1] = pxm * wy;   // bi=0,bj=1
        pxy[2] = wx * pym;   // bi=1,bj=0
        pxy[3] = wx * wy;    // bi=1,bj=1

        // hoist per-dimension index components out of the corner loop
        unsigned int ycomp[2], zcomp[2];
        unsigned int xbase = ix;
        bool useHash;
        if (dense) {
            const unsigned int ur = (unsigned int)res;
            const unsigned int ur2 = ur * ur;
            ycomp[0] = iy * ur;        ycomp[1] = (iy + 1u) * ur;
            zcomp[0] = iz * ur2;       zcomp[1] = (iz + 1u) * ur2;
            useHash = false;
        } else {
            ycomp[0] = iy * 2654435761u;       ycomp[1] = (iy + 1u) * 2654435761u;
            zcomp[0] = iz * 805459861u;        zcomp[1] = (iz + 1u) * 805459861u;
            useHash = true;
        }

        const float2* lt = tbl2 + (size_t)l * TBL_SIZE;
        float f0 = 0.0f, f1 = 0.0f;
        #pragma unroll
        for (int c = 0; c < 8; ++c) {
            const unsigned int bi = (c >> 2) & 1u, bj = (c >> 1) & 1u, bk = c & 1u;
            const unsigned int cx = xbase + bi;
            unsigned int idx;
            if (useHash) idx = (cx ^ ycomp[bj] ^ zcomp[bk]) & TBL_MASK;
            else         idx = cx + ycomp[bj] + zcomp[bk];
            const float cw = pxy[(c >> 1)] * (bk ? wz : pzm);
            const float2 g = lt[idx];
            f0 += cw * g.x;
            f1 += cw * g.y;
        }
        union { _Float16 h[2]; unsigned int u; } pk;
        pk.h[0] = (_Float16)f0;
        pk.h[1] = (_Float16)f1;
        myStageU[m * 36 + l] = pk.u;  // enc columns k' = 2l, 2l+1

        sp *= S2;
    }

    __syncthreads();

    // ---------- layer 1 via WMMA: [16 pts x 64K] @ [64K x 64N] ----------
    // A (16x32 f16): lane hh=0 -> V0..3:K=0..7, V4..7:K=16..23 ; hh=1 -> K=8..15, K=24..31
    const uint4* stq = (const uint4*)myStage;  // 9 uint4 per 72-half row
    Frag A[2];
    #pragma unroll
    for (int kc = 0; kc < 2; ++kc) {
        A[kc].q[0] = stq[m * 9 + kc * 4 + hh];
        A[kc].q[1] = stq[m * 9 + kc * 4 + hh + 2];
    }

    // B (32x16 f16): lane n = nt*16 + m ; lanes 0-15 hold K=0..7,16..23, lanes 16-31 K=8..15,24..31
    const uint4* wq = (const uint4*)w1s;
    const float b2v = b2[0];

    float partial[8];
    #pragma unroll
    for (int r = 0; r < 8; ++r) partial[r] = 0.0f;

    #pragma unroll
    for (int nt = 0; nt < 4; ++nt) {
        const int n = nt * 16 + m;
        Frag B0, B1;
        B0.q[0] = wq[n * 9 + 0 * 4 + hh];
        B0.q[1] = wq[n * 9 + 0 * 4 + hh + 2];
        B1.q[0] = wq[n * 9 + 1 * 4 + hh];
        B1.q[1] = wq[n * 9 + 1 * 4 + hh + 2];

        v8f acc = {};
        acc = __builtin_amdgcn_wmma_f32_16x16x32_f16(false, A[0].h, false, B0.h,
                                                     (short)0, acc, false, false);
        acc = __builtin_amdgcn_wmma_f32_16x16x32_f16(false, A[1].h, false, B1.h,
                                                     (short)0, acc, false, false);

        // D layout: lane (0-15): N=lane, VGPR r -> M=r ; lane (16-31): N=lane-16, M=8+r
        const float bias = b1s[n];
        const float wv   = w2s[n];   // already includes the /100
        #pragma unroll
        for (int r = 0; r < 8; ++r) {
            partial[r] += softplus100_raw(acc[r] + bias) * wv;
        }
    }

    // ---------- layer 2 reduction across the 16 lanes of each half ----------
    #pragma unroll
    for (int r = 0; r < 8; ++r) {
        float v = partial[r];
        v += __shfl_xor(v, 1, 32);
        v += __shfl_xor(v, 2, 32);
        v += __shfl_xor(v, 4, 32);
        v += __shfl_xor(v, 8, 32);
        partial[r] = v;
    }

    if (m == 0) {  // lane 0 (rows 0..7) and lane 16 (rows 8..15)
        #pragma unroll
        for (int r = 0; r < 8; ++r) {
            const long long oi = tileBase + (long long)(hh * 8 + r);
            if (oi < N) out[oi] = partial[r] + b2v;
        }
    }
}

extern "C" void kernel_launch(void* const* d_in, const int* in_sizes, int n_in,
                              void* d_out, int out_size, void* d_ws, size_t ws_size,
                              hipStream_t stream) {
    const float* points = (const float*)d_in[0];
    const float* table  = (const float*)d_in[1];
    const float* v1     = (const float*)d_in[2];
    const float* g1     = (const float*)d_in[3];
    const float* b1     = (const float*)d_in[4];
    const float* v2     = (const float*)d_in[5];
    const float* g2     = (const float*)d_in[6];
    const float* b2     = (const float*)d_in[7];
    float* out = (float*)d_out;

    const long long N = (long long)in_sizes[0] / 3;  // 2,097,152
    const int blocks = (int)((N + 127) / 128);       // 128 points per 256-thread block
    volumesdf_kernel<<<blocks, 256, 0, stream>>>(points, table, v1, g1, b1, v2, g2, b2, out, N);
    (void)n_in; (void)out_size; (void)d_ws; (void)ws_size;
}
